// MPNN_47974784696409
// MI455X (gfx1250) — matrix operand order repaired and checked by reading, not compiled, not run on hardware
//
#include <hip/hip_runtime.h>

namespace {

typedef __attribute__((ext_vector_type(16))) _Float16 v16h;
typedef __attribute__((ext_vector_type(8)))  _Float16 v8h;
typedef __attribute__((ext_vector_type(8)))  float    v8f;

constexpr int N  = 16384;   // nodes
constexpr int E  = 65536;   // edges
constexpr int G  = 64;      // graphs
constexpr int F  = 92;      // node feature dim (padded to 96)
constexpr int FE = 41;      // edge feature dim (padded to 64)
constexpr int H  = 64;      // hidden
constexpr float kInvStd = 0.999500374f;  // 1/sqrt(1+1e-3)

// ---------------------------------------------------------------- helpers
__device__ __forceinline__ v16h mk16(v8h lo, v8h hi) {
  return __builtin_shufflevector(lo, hi, 0,1,2,3,4,5,6,7,8,9,10,11,12,13,14,15);
}
__device__ __forceinline__ v8f wmma16(v16h a, v16h b, v8f c) {
  return __builtin_amdgcn_wmma_f32_16x16x32_f16(false, a, false, b, (short)0, c,
                                                false, false);
}
// A-frag (16x32 f16): lane m=l&15, kh=l>>4; halves 0..15 = K 16kh+0..15 contiguous
__device__ __forceinline__ v16h load_a_frag(const _Float16* row, int kc, int kh) {
  v8h lo = *(const v8h*)(row + kc * 32 + 16 * kh);
  v8h hi = *(const v8h*)(row + kc * 32 + 16 * kh + 8);
  return mk16(lo, hi);
}
// wait for all of this wave's outstanding async (L2->LDS) transfers
__device__ __forceinline__ void wait_async0() {
#if __has_builtin(__builtin_amdgcn_s_wait_asynccnt)
  __builtin_amdgcn_s_wait_asynccnt(0);
#else
  asm volatile("s_wait_asynccnt 0" ::: "memory");
#endif
}
// per-lane 16B async copy global -> LDS (CDNA5 GLOBAL_LOAD_ASYNC_TO_LDS_B128)
__device__ __forceinline__ void async_copy16(uint32_t lds_addr, const void* gptr) {
  asm volatile("global_load_async_to_lds_b128 %0, %1, off"
               :: "v"(lds_addr), "v"(gptr)
               : "memory");
}

// ---------------------------------------------------------------- small kernels
__global__ void zero_f32(float* p, int n) {
  int i = blockIdx.x * 256 + threadIdx.x;
  if (i < n) p[i] = 0.f;
}

__global__ void cvt_pad(const float* __restrict__ src, _Float16* __restrict__ dst,
                        int rows, int Kin, int Kout) {
  int idx = blockIdx.x * 256 + threadIdx.x;
  if (idx >= rows * Kout) return;
  int r = idx / Kout, k = idx - r * Kout;
  dst[idx] = (k < Kin) ? (_Float16)src[(size_t)r * Kin + k] : (_Float16)0.f;
}

// ew_W2 [3][64][4096] f32 -> W2T [3][4096][64] f16
__global__ void transpose_w2(const float* __restrict__ w2, _Float16* __restrict__ w2t) {
  int idx = blockIdx.x * 256 + threadIdx.x;
  if (idx >= 3 * 4096 * 64) return;
  int l = idx / (4096 * 64);
  int rem = idx - l * 4096 * 64;
  int c = rem >> 6, k = rem & 63;
  w2t[idx] = (_Float16)w2[(size_t)l * 64 * 4096 + (size_t)k * 4096 + c];
}

__global__ void count_deg(const int* __restrict__ dst, float* deg) {
  int e = blockIdx.x * 256 + threadIdx.x;
  if (e < E) atomicAdd(&deg[dst[e]], 1.f);
}

__global__ void normalize_agg(const float* __restrict__ agg, const float* __restrict__ deg,
                              _Float16* __restrict__ out) {
  int idx = blockIdx.x * 256 + threadIdx.x;
  if (idx >= N * H) return;
  float dv = fmaxf(deg[idx >> 6], 1.f);
  out[idx] = (_Float16)(agg[idx] / dv);
}

__global__ void gru_gate(const float* __restrict__ xz, const float* __restrict__ hz,
                         float* __restrict__ state, _Float16* __restrict__ state16,
                         float* __restrict__ h, const float* __restrict__ g,
                         const float* __restrict__ be) {
  int idx = blockIdx.x * 256 + threadIdx.x;
  if (idx >= N * H) return;
  int n = idx >> 6, c = idx & 63;
  const float* x  = xz + (size_t)n * 192;
  const float* hr = hz + (size_t)n * 192;
  float z = 1.f / (1.f + __expf(-(x[c] + hr[c])));
  float r = 1.f / (1.f + __expf(-(x[64 + c] + hr[64 + c])));
  float cand = tanhf(x[128 + c] + r * hr[128 + c]);
  float s = state[idx];
  float sn = z * s + (1.f - z) * cand;
  state[idx]   = sn;
  state16[idx] = (_Float16)sn;
  h[idx] = tanhf(g[c] * (sn * kInvStd) + be[c]);
}

__global__ void pool_acc(const float* __restrict__ h, const int* __restrict__ gidx,
                         float* gsum, float* gcnt) {
  int idx = blockIdx.x * 256 + threadIdx.x;
  if (idx >= N * H) return;
  int n = idx >> 6, c = idx & 63;
  int gg = gidx[n];
  atomicAdd(&gsum[gg * H + c], h[idx]);
  if (c == 0) atomicAdd(&gcnt[gg], 1.f);
}

__global__ void pool_fin(const float* __restrict__ gsum, const float* __restrict__ gcnt,
                         _Float16* __restrict__ g16) {
  int idx = blockIdx.x * 256 + threadIdx.x;
  if (idx >= G * H) return;
  g16[idx] = (_Float16)(gsum[idx] / fmaxf(gcnt[idx >> 6], 1.f));
}

__global__ void final_out(const _Float16* __restrict__ r2, const float* __restrict__ W,
                          const float* __restrict__ b, float* __restrict__ out) {
  int gg = threadIdx.x;
  if (gg < G) {
    float acc = b[0];
    for (int c = 0; c < H; ++c) acc += (float)r2[gg * H + c] * W[c];
    out[gg] = acc;
  }
}

// ---------------------------------------------------------------- generic WMMA GEMM
// out[rows,C] = epilogue(A[rows,K]f16 @ W[K_act,C]f32 + bias), per-wave 16-row tile.
// Epilogue: optional bias, optional BN (gamma: g*(x*invstd)+beta), optional relu,
// stores f32 and/or f16.
template <int KCH>  // K = KCH*32 (2 or 3)
__global__ __launch_bounds__(128) void wmma_gemm_k(
    const _Float16* __restrict__ A, int rows,
    const float* __restrict__ W, int K_act, int C,
    const float* __restrict__ bias, const float* __restrict__ gamma,
    const float* __restrict__ beta, int relu,
    float* __restrict__ outF, _Float16* __restrict__ outH) {
  constexpr int K = KCH * 32;
  __shared__ alignas(32) _Float16 WT[192 * 96];  // W transposed: [C][K] f16 (<=36KB)

  int tid = threadIdx.x;
  for (int idx = tid; idx < C * K; idx += 128) {
    int c = idx / K, k = idx - c * K;
    WT[c * K + k] = (k < K_act) ? (_Float16)W[(size_t)k * C + c] : (_Float16)0.f;
  }
  __syncthreads();

  int wave = tid >> 5, lane = tid & 31;
  int n16 = lane & 15, kh = lane >> 4;
  int rt = blockIdx.x * 4 + wave;
  if (rt * 16 >= rows) return;

  const _Float16* Arow = A + (size_t)(rt * 16 + n16) * K;
  v16h af[KCH];
#pragma unroll
  for (int kc = 0; kc < KCH; ++kc) af[kc] = load_a_frag(Arow, kc, kh);

  for (int ct = 0; ct < (C >> 4); ++ct) {
    int col = ct * 16 + n16;
    v8f d = {};
#pragma unroll
    for (int kc = 0; kc < KCH; ++kc) {
      // B-frag (32x16): lane col n=l&15; K = 16*kh + 0..15 contiguous
      v16h bf = *(const v16h*)(&WT[(size_t)col * K + kc * 32 + 16 * kh]);
      d = wmma16(af[kc], bf, d);
    }
    float bv  = bias  ? bias[col]  : 0.f;
    float gv  = gamma ? gamma[col] : 0.f;
    float bev = gamma ? beta[col]  : 0.f;
#pragma unroll
    for (int r = 0; r < 8; ++r) {
      float v = d[r] + bv;
      if (gamma) v = gv * (v * kInvStd) + bev;
      if (relu) v = fmaxf(v, 0.f);
      size_t off = (size_t)(rt * 16 + 8 * kh + r) * C + col;
      if (outF) outF[off] = v;
      if (outH) outH[off] = (_Float16)v;
    }
  }
}

// ---------------------------------------------------------------- fused message kernel
// Per 16-edge tile (1 wave): D'[c,edge] = W2T[c,:] @ t[edge,:]^T via WMMA,
// msgs[edge,i] = sum_c (D'[c,edge]+b2[c]) * h[src[edge], c&63], atomically scattered
// into agg[dst]. The W2T row-block for each output dim i is staged in LDS (shared by
// all 8 waves) with DOUBLE-BUFFERED async L2->LDS copies (ASYNCcnt) so the staging
// latency hides behind the WMMA work of the current iteration.
__global__ __launch_bounds__(256) void msg_kernel(
    const _Float16* __restrict__ t16,   // [E,64] f16
    const _Float16* __restrict__ W2T,   // [4096,64] f16 (layer slice)
    const float* __restrict__ b2,       // [4096]   (layer slice)
    const float* __restrict__ h,        // [N,64] f32
    const int* __restrict__ src_idx, const int* __restrict__ dst_idx,
    float* __restrict__ agg) {          // [N,64] f32 accumulator
  __shared__ alignas(32) _Float16 Ai[2][64 * 64];  // ping-pong W2T row blocks (16KB)
  __shared__ float b2s[2][64];
  __shared__ float hs[8][16][64];                  // gathered source features (32KB)

  int tid = threadIdx.x, wave = tid >> 5, lane = tid & 31;
  int n16 = lane & 15, kh = lane >> 4;
  int eg = (blockIdx.x * 8 + wave) * 16 + n16;
  int s = src_idx[eg];
  int dnode = dst_idx[eg];

  {  // gather h[src] tile into LDS: lane copies 32 consecutive floats of its edge row
    const float4* hp = (const float4*)(h + (size_t)s * H + kh * 32);
    float4* op = (float4*)(&hs[wave][n16][kh * 32]);
#pragma unroll
    for (int i = 0; i < 8; ++i) op[i] = hp[i];
  }
  // B-frags: t^T for this wave's 16 edges, reused for all 64 output dims
  const _Float16* trow = t16 + (size_t)eg * 64;
  v16h bf0 = *(const v16h*)(trow + 16 * kh);
  v16h bf1 = *(const v16h*)(trow + 32 + 16 * kh);

  // async staging addresses: each of 256 threads moves 2 x 16B per 8KB buffer;
  // copy u covers the contiguous 4KB span [u*4096, u*4096+4096), thread at tid*16.
  uint32_t ldsA[2] = {(uint32_t)(uintptr_t)(&Ai[0][tid * 8]),
                      (uint32_t)(uintptr_t)(&Ai[1][tid * 8])};
  const char* gsrc0 = (const char*)W2T + (size_t)tid * 16;

  auto issue = [&](int i) {
    uint32_t dst = ldsA[i & 1];
    const char* src = gsrc0 + (size_t)i * 8192;
#pragma unroll
    for (int u = 0; u < 2; ++u)
      async_copy16(dst + (uint32_t)u * 4096u, src + (size_t)u * 4096);
  };

  issue(0);
  if (tid < 64) b2s[0][tid] = b2[tid];

  for (int i = 0; i < 64; ++i) {
    int p = i & 1;
    wait_async0();     // this wave's fills of buffer p are complete
    __syncthreads();   // all waves' fills done; previous buffer fully consumed
    if (i + 1 < 64) {  // start filling the other buffer while we compute
      issue(i + 1);
      if (tid < 64) b2s[1 - p][tid] = b2[(i + 1) * 64 + tid];
    }

    float acc = 0.f;
#pragma unroll
    for (int q = 0; q < 4; ++q) {
      const _Float16* arow = &Ai[p][(16 * q + n16) * 64];
      v8f d = {};
      d = wmma16(load_a_frag(arow, 0, kh), bf0, d);
      d = wmma16(load_a_frag(arow, 1, kh), bf1, d);
      int jb = 16 * q + 8 * kh;  // lane's j = column index within H
      const float* hrow = &hs[wave][n16][0];
#pragma unroll
      for (int r = 0; r < 8; ++r) acc += (d[r] + b2s[p][jb + r]) * hrow[jb + r];
    }
    acc += __shfl_xor(acc, 16, 32);  // combine the two K-half lanes
    if (lane < 16) atomicAdd(&agg[(size_t)dnode * H + i], acc);
  }
}

}  // namespace

// ---------------------------------------------------------------- host orchestration
extern "C" void kernel_launch(void* const* d_in, const int* in_sizes, int n_in,
                              void* d_out, int out_size, void* d_ws, size_t ws_size,
                              hipStream_t stream) {
  (void)in_sizes; (void)n_in; (void)out_size; (void)ws_size;
  const float* nf     = (const float*)d_in[0];
  const float* ef     = (const float*)d_in[1];
  const int*   eidx   = (const int*)d_in[2];
  const int*   gidx   = (const int*)d_in[3];
  const float* ne_W   = (const float*)d_in[4];
  const float* ne_b   = (const float*)d_in[5];
  const float* ne_g   = (const float*)d_in[6];
  const float* ne_be  = (const float*)d_in[7];
  const float* ee_W   = (const float*)d_in[8];
  const float* ee_b   = (const float*)d_in[9];
  const float* ee_g   = (const float*)d_in[10];
  const float* ee_be  = (const float*)d_in[11];
  const float* ew_W1  = (const float*)d_in[12];
  const float* ew_b1  = (const float*)d_in[13];
  const float* ew_W2  = (const float*)d_in[14];
  const float* ew_b2  = (const float*)d_in[15];
  const float* gru_Wx = (const float*)d_in[16];
  const float* gru_Wh = (const float*)d_in[17];
  const float* gru_bin  = (const float*)d_in[18];
  const float* gru_brec = (const float*)d_in[19];
  const float* mb_g   = (const float*)d_in[20];
  const float* mb_be  = (const float*)d_in[21];
  const float* ro_W   = (const float*)d_in[22];
  const float* ro_b   = (const float*)d_in[23];
  const float* ro_g   = (const float*)d_in[24];
  const float* ro_be  = (const float*)d_in[25];
  const float* out_W  = (const float*)d_in[26];
  const float* out_b  = (const float*)d_in[27];
  const int* src_idx = eidx;
  const int* dst_idx = eidx + E;

  // workspace carve (256B aligned)
  char* ws = (char*)d_ws;
  size_t off = 0;
  auto carve = [&](size_t bytes) -> char* {
    char* p = ws + off;
    off += (bytes + 255) & ~(size_t)255;
    return p;
  };
  float*    hbuf    = (float*)   carve((size_t)N * H * 4);
  float*    state   = (float*)   carve((size_t)N * H * 4);
  _Float16* state16 = (_Float16*)carve((size_t)N * H * 2);
  _Float16* nf16    = (_Float16*)carve((size_t)N * 96 * 2);
  _Float16* efeat16 = (_Float16*)carve((size_t)E * 64 * 2);
  _Float16* e16     = (_Float16*)carve((size_t)E * H * 2);
  _Float16* t16     = (_Float16*)carve((size_t)E * H * 2);
  _Float16* W2T     = (_Float16*)carve((size_t)3 * 4096 * 64 * 2);
  float*    agg     = (float*)   carve((size_t)N * H * 4);
  _Float16* aggn16  = (_Float16*)carve((size_t)N * H * 2);
  float*    xz      = (float*)   carve((size_t)N * 192 * 4);
  float*    hz      = (float*)   carve((size_t)N * 192 * 4);
  float*    deg     = (float*)   carve((size_t)N * 4);
  float*    gsum    = (float*)   carve((size_t)G * H * 4);
  float*    gcnt    = (float*)   carve((size_t)G * 4);
  _Float16* g16     = (_Float16*)carve((size_t)G * H * 2);
  _Float16* r16a    = (_Float16*)carve((size_t)G * H * 2);
  _Float16* r16b    = (_Float16*)carve((size_t)G * H * 2);

  auto zf = [&](float* p, int n) {
    zero_f32<<<(n + 255) / 256, 256, 0, stream>>>(p, n);
  };
  auto gemm = [&](const _Float16* A, int rows, int Kp, const float* W, int K_act,
                  int C, const float* bias, const float* gamma, const float* beta,
                  int relu, float* outF, _Float16* outH) {
    if (Kp == 96)
      wmma_gemm_k<3><<<rows / 64, 128, 0, stream>>>(A, rows, W, K_act, C, bias,
                                                    gamma, beta, relu, outF, outH);
    else
      wmma_gemm_k<2><<<rows / 64, 128, 0, stream>>>(A, rows, W, K_act, C, bias,
                                                    gamma, beta, relu, outF, outH);
  };

  // preprocessing: f16 conversions, W2 transpose, degrees
  cvt_pad<<<(N * 96 + 255) / 256, 256, 0, stream>>>(nf, nf16, N, F, 96);
  cvt_pad<<<(E * 64 + 255) / 256, 256, 0, stream>>>(ef, efeat16, E, FE, 64);
  transpose_w2<<<(3 * 4096 * 64 + 255) / 256, 256, 0, stream>>>(ew_W2, W2T);
  zf(deg, N);
  count_deg<<<E / 256, 256, 0, stream>>>(dst_idx, deg);

  // embeddings: Dense -> BN -> relu
  gemm(nf16, N, 96, ne_W, F, H, ne_b, ne_g, ne_be, 1, hbuf, state16);
  hipMemcpyAsync(state, hbuf, (size_t)N * H * 4, hipMemcpyDeviceToDevice, stream);
  gemm(efeat16, E, 64, ee_W, FE, H, ee_b, ee_g, ee_be, 1, nullptr, e16);

  // message passing layers
  for (int l = 0; l < 3; ++l) {
    zf(agg, N * H);
    gemm(e16, E, 64, ew_W1 + l * H * H, H, H, ew_b1 + l * H, nullptr, nullptr, 1,
         nullptr, t16);
    msg_kernel<<<E / 128, 256, 0, stream>>>(t16, W2T + (size_t)l * 4096 * 64,
                                            ew_b2 + l * 4096, hbuf, src_idx, dst_idx,
                                            agg);
    normalize_agg<<<(N * H + 255) / 256, 256, 0, stream>>>(agg, deg, aggn16);
    gemm(aggn16, N, 64, gru_Wx + l * H * 3 * H, H, 3 * H, gru_bin + l * 3 * H,
         nullptr, nullptr, 0, xz, nullptr);
    gemm(state16, N, 64, gru_Wh + l * H * 3 * H, H, 3 * H, gru_brec + l * 3 * H,
         nullptr, nullptr, 0, hz, nullptr);
    gru_gate<<<(N * H + 255) / 256, 256, 0, stream>>>(xz, hz, state, state16, hbuf,
                                                      mb_g + l * H, mb_be + l * H);
  }

  // graph pooling + readout
  zf(gsum, G * H);
  zf(gcnt, G);
  pool_acc<<<(N * H + 255) / 256, 256, 0, stream>>>(hbuf, gidx, gsum, gcnt);
  pool_fin<<<(G * H + 255) / 256, 256, 0, stream>>>(gsum, gcnt, g16);
  gemm(g16, G, 64, ro_W, H, H, ro_b, ro_g, ro_be, 1, nullptr, r16a);
  gemm(r16a, G, 64, ro_W + H * H, H, H, ro_b + H, ro_g + H, ro_be + H, 1, nullptr,
       r16b);
  final_out<<<1, 64, 0, stream>>>(r16b, out_W, out_b, (float*)d_out);
}
